// _GraphormerLayer_48455821033931
// MI455X (gfx1250) — matrix operand order repaired and checked by reading, not compiled, not run on hardware
//
#include <hip/hip_runtime.h>
#include <hip/hip_bf16.h>

#define B_   16
#define L_   512
#define HID_ 768
#define NH_  12
#define HD_  64
#define FFN_ 3072
#define M_   (B_*L_)   // 8192

typedef __attribute__((ext_vector_type(16))) __bf16 v16bf;
typedef __attribute__((ext_vector_type(8)))  __bf16 v8bf;
typedef __attribute__((ext_vector_type(8)))  float  v8f;

// round-to-nearest-even f32 -> bf16
static __device__ inline __bf16 f2bf(float f) {
  unsigned u = __builtin_bit_cast(unsigned, f);
  unsigned r = (u + 0x7FFFu + ((u >> 16) & 1u)) >> 16;
  unsigned short s = (unsigned short)r;
  return __builtin_bit_cast(__bf16, s);
}

static __device__ inline v8f wmma_bf16(v8bf alo, v8bf ahi, v8bf blo, v8bf bhi, v8f c) {
  v16bf a, b;
#pragma unroll
  for (int i = 0; i < 8; ++i) { a[i] = alo[i]; a[i + 8] = ahi[i]; b[i] = blo[i]; b[i + 8] = bhi[i]; }
  return __builtin_amdgcn_wmma_f32_16x16x32_bf16(false, a, false, b, (short)0, c, false, false);
}

// ---------------- weight transpose + f32->bf16 :  WT[n*K + k] = W[k*N + n]
__global__ void k_w_transpose(const float* __restrict__ W, __bf16* __restrict__ WT, int K, int N) {
  __shared__ float t[32][33];
  int n0 = blockIdx.x * 32, k0 = blockIdx.y * 32;
  int tx = threadIdx.x, ty = threadIdx.y;      // 32 x 8
#pragma unroll
  for (int i = 0; i < 32; i += 8)
    t[ty + i][tx] = W[(size_t)(k0 + ty + i) * N + (n0 + tx)];
  __syncthreads();
#pragma unroll
  for (int i = 0; i < 32; i += 8)
    WT[(size_t)(n0 + ty + i) * K + (k0 + tx)] = f2bf(t[tx][ty + i]);
}

// ---------------- layernorm: f32 in -> bf16 out (one row per block, 256 thr)
__global__ void k_layernorm_bf16(const float* __restrict__ x, const float* __restrict__ g,
                                 const float* __restrict__ b, __bf16* __restrict__ out) {
  int row = blockIdx.x, tid = threadIdx.x;
  const float* xr = x + (size_t)row * HID_;
  float v0 = xr[tid], v1 = xr[tid + 256], v2 = xr[tid + 512];
  float s = v0 + v1 + v2, s2 = v0 * v0 + v1 * v1 + v2 * v2;
#pragma unroll
  for (int m = 16; m >= 1; m >>= 1) { s += __shfl_xor(s, m, 32); s2 += __shfl_xor(s2, m, 32); }
  __shared__ float rs[8], rs2[8], mv[2];
  int wid = tid >> 5;
  if ((tid & 31) == 0) { rs[wid] = s; rs2[wid] = s2; }
  __syncthreads();
  if (tid == 0) {
    float ts = 0.f, ts2 = 0.f;
#pragma unroll
    for (int i = 0; i < 8; ++i) { ts += rs[i]; ts2 += rs2[i]; }
    float mu = ts / HID_;
    float var = ts2 / HID_ - mu * mu;
    mv[0] = mu; mv[1] = rsqrtf(var + 1e-5f);
  }
  __syncthreads();
  float mu = mv[0], inv = mv[1];
  __bf16* orow = out + (size_t)row * HID_;
#pragma unroll
  for (int i = 0; i < 3; ++i) {
    int c = tid + i * 256;
    orow[c] = f2bf((xr[c] - mu) * inv * g[c] + b[c]);
  }
}

// ---------------- bf16 WMMA GEMM, 128x64 block tile, 32x32 per wave, pipelined
#define EP_BF16  0   // bf16 store
#define EP_VT    1   // bf16 store, V transposed per head -> [b,h,d,l]
#define EP_RESID 2   // f32 store: acc + bias + resid
#define EP_GELU  3   // bf16 store of exact gelu(acc+bias)

#define EPILOGUE(acc, rt, ct)                                                  \
  {                                                                            \
    _Pragma("unroll")                                                          \
    for (int j = 0; j < 8; ++j) {                                              \
      int m = mw + (rt) * 16 + j + hf * 8;                                     \
      int n = nw + (ct) * 16 + l16;                                            \
      float v = acc[j] + bias[n];                                              \
      size_t idx = (size_t)m * Ndim + n;                                       \
      if (mode == EP_BF16) {                                                   \
        outb[idx] = f2bf(v);                                                   \
      } else if (mode == EP_VT) {                                              \
        int bb = m >> 9, ll = m & 511, hh = n >> 6, dd = n & 63;               \
        outb[((size_t)(bb * NH_ + hh) * HD_ + dd) * L_ + ll] = f2bf(v);        \
      } else if (mode == EP_RESID) {                                           \
        outf[idx] = v + resid[idx];                                            \
      } else {                                                                 \
        outb[idx] = f2bf(0.5f * v * (1.0f + erff(v * 0.70710678118f)));        \
      }                                                                        \
    }                                                                          \
  }

__global__ void k_gemm_bf16(const __bf16* __restrict__ A, const __bf16* __restrict__ WT,
                            const float* __restrict__ bias, const float* __restrict__ resid,
                            __bf16* __restrict__ outb, float* __restrict__ outf,
                            int Ndim, int Kdim, int mode) {
  int lane = threadIdx.x & 31, w = threadIdx.x >> 5;
  int hf = lane >> 4, l16 = lane & 15;
  int aOff = hf * 8, bOff = hf * 16;
  int m0 = blockIdx.y * 128, n0 = blockIdx.x * 64;
  int wr = w >> 1, wc = w & 1;                 // 4x2 waves of 32x32 tiles
  int mw = m0 + wr * 32, nw = n0 + wc * 32;

  const __bf16* Ar0 = A  + (size_t)(mw + l16) * Kdim;
  const __bf16* Ar1 = A  + (size_t)(mw + 16 + l16) * Kdim;
  const __bf16* Br0 = WT + (size_t)(nw + l16) * Kdim;
  const __bf16* Br1 = WT + (size_t)(nw + 16 + l16) * Kdim;

  v8f acc00 = {}, acc01 = {}, acc10 = {}, acc11 = {};

  // preload fragments for k = 0
  v8bf a0lo = *(const v8bf*)(Ar0 + aOff);
  v8bf a0hi = *(const v8bf*)(Ar0 + 16 + aOff);
  v8bf a1lo = *(const v8bf*)(Ar1 + aOff);
  v8bf a1hi = *(const v8bf*)(Ar1 + 16 + aOff);
  v8bf b0lo = *(const v8bf*)(Br0 + aOff);
  v8bf b0hi = *(const v8bf*)(Br0 + 16 + aOff);
  v8bf b1lo = *(const v8bf*)(Br1 + aOff);
  v8bf b1hi = *(const v8bf*)(Br1 + 16 + aOff);

  for (int k = 32; k < Kdim; k += 32) {
    // issue next-step loads before consuming current fragments
    v8bf na0lo = *(const v8bf*)(Ar0 + k + aOff);
    v8bf na0hi = *(const v8bf*)(Ar0 + k + 16 + aOff);
    v8bf na1lo = *(const v8bf*)(Ar1 + k + aOff);
    v8bf na1hi = *(const v8bf*)(Ar1 + k + 16 + aOff);
    v8bf nb0lo = *(const v8bf*)(Br0 + k + aOff);
    v8bf nb0hi = *(const v8bf*)(Br0 + k + 16 + aOff);
    v8bf nb1lo = *(const v8bf*)(Br1 + k + aOff);
    v8bf nb1hi = *(const v8bf*)(Br1 + k + 16 + aOff);

    acc00 = wmma_bf16(a0lo, a0hi, b0lo, b0hi, acc00);
    acc01 = wmma_bf16(a0lo, a0hi, b1lo, b1hi, acc01);
    acc10 = wmma_bf16(a1lo, a1hi, b0lo, b0hi, acc10);
    acc11 = wmma_bf16(a1lo, a1hi, b1lo, b1hi, acc11);

    a0lo = na0lo; a0hi = na0hi; a1lo = na1lo; a1hi = na1hi;
    b0lo = nb0lo; b0hi = nb0hi; b1lo = nb1lo; b1hi = nb1hi;
  }
  acc00 = wmma_bf16(a0lo, a0hi, b0lo, b0hi, acc00);
  acc01 = wmma_bf16(a0lo, a0hi, b1lo, b1hi, acc01);
  acc10 = wmma_bf16(a1lo, a1hi, b0lo, b0hi, acc10);
  acc11 = wmma_bf16(a1lo, a1hi, b1lo, b1hi, acc11);

  EPILOGUE(acc00, 0, 0)
  EPILOGUE(acc01, 0, 1)
  EPILOGUE(acc10, 1, 0)
  EPILOGUE(acc11, 1, 1)
}

// ---------------- fused attention: scores -> softmax -> context, per (b*h, 32 query rows)
__global__ void k_attention(const __bf16* __restrict__ qb, const __bf16* __restrict__ kb,
                            const __bf16* __restrict__ vt, const float* __restrict__ abias,
                            const unsigned char* __restrict__ mask,
                            float* __restrict__ attn_w, __bf16* __restrict__ ctx) {
  __shared__ float sc[32][512];        // 64 KB
  int bh = blockIdx.x;
  int b = bh / NH_, h = bh % NH_;
  int r0 = blockIdx.y * 32;
  int lane = threadIdx.x & 31, w = threadIdx.x >> 5;
  int hf = lane >> 4, l16 = lane & 15;
  int aOff = hf * 8, bOff = hf * 16;

  // phase 1: scores = Q K^T / 8 + bias (mask). Wave holds its Q fragments in
  // registers (both K-steps) and sweeps 8 key tiles with them.
  {
    int rt = w & 1;
    int ntBase = (w >> 1) * 8;
    int lq = r0 + rt * 16 + l16;
    const __bf16* Qr = qb + ((size_t)b * L_ + lq) * HID_ + h * HD_;
    v8bf q0lo = *(const v8bf*)(Qr + aOff);
    v8bf q0hi = *(const v8bf*)(Qr + 16 + aOff);
    v8bf q1lo = *(const v8bf*)(Qr + 32 + aOff);
    v8bf q1hi = *(const v8bf*)(Qr + 48 + aOff);
#pragma unroll
    for (int nt = ntBase; nt < ntBase + 8; ++nt) {
      int lk = nt * 16 + l16;
      const __bf16* Kr = kb + ((size_t)b * L_ + lk) * HID_ + h * HD_;
      v8bf k0lo = *(const v8bf*)(Kr + aOff);
      v8bf k0hi = *(const v8bf*)(Kr + 16 + aOff);
      v8bf k1lo = *(const v8bf*)(Kr + 32 + aOff);
      v8bf k1hi = *(const v8bf*)(Kr + 48 + aOff);
      v8f acc = {};
      acc = wmma_bf16(q0lo, q0hi, k0lo, k0hi, acc);
      acc = wmma_bf16(q1lo, q1hi, k1lo, k1hi, acc);
#pragma unroll
      for (int j = 0; j < 8; ++j) {
        int rl = rt * 16 + j + hf * 8;
        int col = nt * 16 + l16;
        float v = acc[j] * 0.125f + abias[(((size_t)bh) * L_ + (r0 + rl)) * L_ + col];
        if (!mask[(size_t)b * L_ + col]) v = -INFINITY;
        sc[rl][col] = v;
      }
    }
  }
  __syncthreads();

  // softmax: 4 rows per wave, shfl_xor reductions; write probs to LDS + d_out
  size_t aw_base = (size_t)bh * L_ * L_;
#pragma unroll
  for (int rr = 0; rr < 4; ++rr) {
    int rl = w * 4 + rr;
    float mx = -INFINITY;
    for (int c = lane; c < 512; c += 32) mx = fmaxf(mx, sc[rl][c]);
#pragma unroll
    for (int m = 16; m >= 1; m >>= 1) mx = fmaxf(mx, __shfl_xor(mx, m, 32));
    float sum = 0.f;
    for (int c = lane; c < 512; c += 32) sum += __expf(sc[rl][c] - mx);
#pragma unroll
    for (int m = 16; m >= 1; m >>= 1) sum += __shfl_xor(sum, m, 32);
    float inv = 1.0f / sum;
    for (int c = lane; c < 512; c += 32) {
      float p = __expf(sc[rl][c] - mx) * inv;
      sc[rl][c] = p;
      attn_w[aw_base + (size_t)(r0 + rl) * L_ + c] = p;
    }
  }
  __syncthreads();

  // phase 2: ctx = P @ V ; P(f32, LDS) -> bf16 frags; V from vt[bh][d][l] contiguous
  {
    int rt = w >> 2, nt = w & 3;
    int rl = rt * 16 + l16;
    int d  = nt * 16 + l16;
    const __bf16* Vr = vt + ((size_t)bh * HD_ + d) * L_;
    v8f acc = {};
    for (int k = 0; k < 512; k += 32) {
      const float4* p0 = (const float4*)&sc[rl][k + aOff];
      const float4* p1 = (const float4*)&sc[rl][k + 16 + aOff];
      float4 f0 = p0[0], f1 = p0[1], f2v = p1[0], f3 = p1[1];
      v8bf alo, ahi;
      alo[0] = f2bf(f0.x); alo[1] = f2bf(f0.y); alo[2] = f2bf(f0.z); alo[3] = f2bf(f0.w);
      alo[4] = f2bf(f1.x); alo[5] = f2bf(f1.y); alo[6] = f2bf(f1.z); alo[7] = f2bf(f1.w);
      ahi[0] = f2bf(f2v.x); ahi[1] = f2bf(f2v.y); ahi[2] = f2bf(f2v.z); ahi[3] = f2bf(f2v.w);
      ahi[4] = f2bf(f3.x);  ahi[5] = f2bf(f3.y);  ahi[6] = f2bf(f3.z);  ahi[7] = f2bf(f3.w);
      v8bf blo = *(const v8bf*)(Vr + k + aOff);
      v8bf bhi = *(const v8bf*)(Vr + k + 16 + aOff);
      acc = wmma_bf16(alo, ahi, blo, bhi, acc);
    }
#pragma unroll
    for (int j = 0; j < 8; ++j) {
      int rl2 = rt * 16 + j + hf * 8;
      int l = r0 + rl2;
      int dd = nt * 16 + l16;
      ctx[((size_t)b * L_ + l) * HID_ + h * HD_ + dd] = f2bf(acc[j]);
    }
  }
}

extern "C" void kernel_launch(void* const* d_in, const int* in_sizes, int n_in,
                              void* d_out, int out_size, void* d_ws, size_t ws_size,
                              hipStream_t stream) {
  const float* x         = (const float*)d_in[0];
  const float* attn_bias = (const float*)d_in[1];
  const unsigned char* mask = (const unsigned char*)d_in[2];
  const float* wq = (const float*)d_in[3];  const float* bq = (const float*)d_in[4];
  const float* wk = (const float*)d_in[5];  const float* bk = (const float*)d_in[6];
  const float* wv = (const float*)d_in[7];  const float* bv = (const float*)d_in[8];
  const float* wo = (const float*)d_in[9];  const float* bo = (const float*)d_in[10];
  const float* ln1_g = (const float*)d_in[11]; const float* ln1_b = (const float*)d_in[12];
  const float* ln2_g = (const float*)d_in[13]; const float* ln2_b = (const float*)d_in[14];
  const float* w1 = (const float*)d_in[15]; const float* b1 = (const float*)d_in[16];
  const float* w2 = (const float*)d_in[17]; const float* b2 = (const float*)d_in[18];

  float* xout   = (float*)d_out;
  float* attn_w = (float*)d_out + (size_t)M_ * HID_;

  char* ws = (char*)d_ws;
  size_t off = 0;
  auto alloc = [&](size_t bytes) -> char* {
    char* p = ws + off; off += (bytes + 255) & ~(size_t)255; return p;
  };
  __bf16* h1   = (__bf16*)alloc((size_t)M_ * HID_ * 2);
  __bf16* wqT  = (__bf16*)alloc((size_t)HID_ * HID_ * 2);
  __bf16* wkT  = (__bf16*)alloc((size_t)HID_ * HID_ * 2);
  __bf16* wvT  = (__bf16*)alloc((size_t)HID_ * HID_ * 2);
  __bf16* woT  = (__bf16*)alloc((size_t)HID_ * HID_ * 2);
  __bf16* w1T  = (__bf16*)alloc((size_t)FFN_ * HID_ * 2);
  __bf16* w2T  = (__bf16*)alloc((size_t)HID_ * FFN_ * 2);
  __bf16* qbuf = (__bf16*)alloc((size_t)M_ * HID_ * 2);
  __bf16* kbuf = (__bf16*)alloc((size_t)M_ * HID_ * 2);
  __bf16* vt   = (__bf16*)alloc((size_t)M_ * HID_ * 2);
  __bf16* ctx  = (__bf16*)alloc((size_t)M_ * HID_ * 2);
  float*  x1   = (float*) alloc((size_t)M_ * HID_ * 4);
  __bf16* h2   = (__bf16*)alloc((size_t)M_ * HID_ * 2);
  __bf16* mid  = (__bf16*)alloc((size_t)M_ * FFN_ * 2);

  dim3 tb(32, 8);
  k_w_transpose<<<dim3(HID_/32, HID_/32), tb, 0, stream>>>(wq, wqT, HID_, HID_);
  k_w_transpose<<<dim3(HID_/32, HID_/32), tb, 0, stream>>>(wk, wkT, HID_, HID_);
  k_w_transpose<<<dim3(HID_/32, HID_/32), tb, 0, stream>>>(wv, wvT, HID_, HID_);
  k_w_transpose<<<dim3(HID_/32, HID_/32), tb, 0, stream>>>(wo, woT, HID_, HID_);
  k_w_transpose<<<dim3(FFN_/32, HID_/32), tb, 0, stream>>>(w1, w1T, HID_, FFN_);
  k_w_transpose<<<dim3(HID_/32, FFN_/32), tb, 0, stream>>>(w2, w2T, FFN_, HID_);

  k_layernorm_bf16<<<M_, 256, 0, stream>>>(x, ln1_g, ln1_b, h1);

  k_gemm_bf16<<<dim3(HID_/64, M_/128), 256, 0, stream>>>(h1, wqT, bq, nullptr, qbuf, nullptr, HID_, HID_, EP_BF16);
  k_gemm_bf16<<<dim3(HID_/64, M_/128), 256, 0, stream>>>(h1, wkT, bk, nullptr, kbuf, nullptr, HID_, HID_, EP_BF16);
  k_gemm_bf16<<<dim3(HID_/64, M_/128), 256, 0, stream>>>(h1, wvT, bv, nullptr, vt,   nullptr, HID_, HID_, EP_VT);

  k_attention<<<dim3(B_*NH_, L_/32), 256, 0, stream>>>(qbuf, kbuf, vt, attn_bias, mask, attn_w, ctx);

  k_gemm_bf16<<<dim3(HID_/64, M_/128), 256, 0, stream>>>(ctx, woT, bo, x, nullptr, x1, HID_, HID_, EP_RESID);

  k_layernorm_bf16<<<M_, 256, 0, stream>>>(x1, ln2_g, ln2_b, h2);

  k_gemm_bf16<<<dim3(FFN_/64, M_/128), 256, 0, stream>>>(h2, w1T, b1, nullptr, mid, nullptr, FFN_, HID_, EP_GELU);

  k_gemm_bf16<<<dim3(HID_/64, M_/128), 256, 0, stream>>>(mid, w2T, b2, x1, nullptr, xout, HID_, FFN_, EP_RESID);
}
